// Qwen3MoeAfdMlpLayer_6073083756731
// MI455X (gfx1250) — matrix operand order repaired and checked by reading, not compiled, not run on hardware
//
#include <hip/hip_runtime.h>

// Problem constants (match reference)
#define TT   1024
#define HH   2048
#define FF   768
#define EE   8
#define TOPK 8

typedef __bf16 bf16_t;
typedef bf16_t v16bf __attribute__((ext_vector_type(16)));
typedef float  v8f   __attribute__((ext_vector_type(8)));

union FragBF { v16bf v; unsigned u[8]; };

// Pack two f32 -> two bf16 (round-to-nearest-even) into one dword.
__device__ __forceinline__ unsigned pack_bf2(float a, float b) {
  unsigned ua = __float_as_uint(a), ub = __float_as_uint(b);
  ua = (ua + 0x7FFFu + ((ua >> 16) & 1u)) >> 16;
  ub = (ub + 0x7FFFu + ((ub >> 16) & 1u)) >> 16;
  return (ua & 0xFFFFu) | (ub << 16);
}

// ---------------------------------------------------------------------------
// Kernel 0: routing weights  rw[t][e] = sum_k tw[t,k] * (ids[t,k]==e)
// ---------------------------------------------------------------------------
__global__ void moe_route(const float* __restrict__ tw,
                          const int*   __restrict__ ids,
                          float*       __restrict__ rw) {
  int t = blockIdx.x * blockDim.x + threadIdx.x;
  if (t >= TT) return;
  float a[EE];
#pragma unroll
  for (int e = 0; e < EE; ++e) a[e] = 0.f;
#pragma unroll
  for (int k = 0; k < TOPK; ++k) {
    int e = ids[t * TOPK + k];
    a[e] += tw[t * TOPK + k];
  }
#pragma unroll
  for (int e = 0; e < EE; ++e) rw[t * EE + e] = a[e];
}

// ---------------------------------------------------------------------------
// Kernel 1: one-pass f32 -> bf16 conversion (each element converted ONCE)
// ---------------------------------------------------------------------------
__global__ __launch_bounds__(256) void cvt_f32_bf16_v4(
    const float* __restrict__ src, unsigned short* __restrict__ dst, int n4) {
  int i = blockIdx.x * blockDim.x + threadIdx.x;
  if (i >= n4) return;
  float4 f = ((const float4*)src)[i];
  uint2 p;
  p.x = pack_bf2(f.x, f.y);
  p.y = pack_bf2(f.z, f.w);
  ((uint2*)dst)[i] = p;
}

// ---------------------------------------------------------------------------
// Kernel 2: GEMM1 + SiGLU, all-bf16 operands.
// Block 256 thr / 8 waves, tile 128(M) x 64(F); waves 0-3 gate, 4-7 up.
// Per wave per k-step: 2 A-frags x 4 B-frags -> 8 WMMA.
// ---------------------------------------------------------------------------
__global__ __launch_bounds__(256) void moe_gemm1_silu(
    const unsigned short* __restrict__ Xb,
    const unsigned short* __restrict__ w1b,
    unsigned short* __restrict__ hbuf) {
  __shared__ __align__(16) unsigned short shA [128 * 40];
  __shared__ __align__(16) unsigned short shBg[64 * 40];
  __shared__ __align__(16) unsigned short shBu[64 * 40];
  __shared__ __align__(16) float          shX [128 * 64];

  const int tid  = threadIdx.x;
  const int wave = tid >> 5, ln = tid & 31, hlf = ln >> 4, nl = ln & 15;
  const int mBase = blockIdx.x * 128, fBase = blockIdx.y * 64, e = blockIdx.z;
  const int m_blk = wave & 3, is_up = wave >> 2;

  const unsigned short* Ap = Xb  + (size_t)mBase * HH;
  const unsigned short* Gp = w1b + ((size_t)e * 2 * FF + fBase) * HH;
  const unsigned short* Up = Gp + (size_t)FF * HH;

  const int lrow = tid >> 2, lq = tid & 3;   // tile-fill coords (8 bf16 each)

  v8f acc[8] = {};

  // register prefetch of k-step 0
  uint4 ra0 = *(const uint4*)(Ap + (size_t)lrow * HH + lq * 8);
  uint4 ra1 = *(const uint4*)(Ap + (size_t)(lrow + 64) * HH + lq * 8);
  uint4 rg  = *(const uint4*)(Gp + (size_t)lrow * HH + lq * 8);
  uint4 ru  = *(const uint4*)(Up + (size_t)lrow * HH + lq * 8);

  for (int kt = 0; kt < HH / 32; ++kt) {
    *(uint4*)&shA [lrow * 40 + lq * 8]        = ra0;
    *(uint4*)&shA [(lrow + 64) * 40 + lq * 8] = ra1;
    *(uint4*)&shBg[lrow * 40 + lq * 8]        = rg;
    *(uint4*)&shBu[lrow * 40 + lq * 8]        = ru;
    __syncthreads();

    if (kt + 1 < HH / 32) {  // overlap next global fetch with WMMAs
      const int kB = (kt + 1) * 32;
      ra0 = *(const uint4*)(Ap + (size_t)lrow * HH + kB + lq * 8);
      ra1 = *(const uint4*)(Ap + (size_t)(lrow + 64) * HH + kB + lq * 8);
      rg  = *(const uint4*)(Gp + (size_t)lrow * HH + kB + lq * 8);
      ru  = *(const uint4*)(Up + (size_t)lrow * HH + kB + lq * 8);
    }

    FragBF af[2];
#pragma unroll
    for (int s = 0; s < 2; ++s) {
      const unsigned short* arow = &shA[((m_blk + 4 * s) * 16 + nl) * 40];
#pragma unroll
      for (int v = 0; v < 8; ++v) {
        int k0 = hlf * 16 + v * 2;
        af[s].u[v] = *(const unsigned*)(arow + k0);
      }
    }
    const unsigned short* bt = is_up ? shBu : shBg;
#pragma unroll
    for (int j = 0; j < 4; ++j) {
      FragBF bfr;
      const unsigned short* brw = &bt[(j * 16 + nl) * 40 + hlf * 16];
#pragma unroll
      for (int v = 0; v < 8; ++v) bfr.u[v] = *(const unsigned*)(brw + 2 * v);
#pragma unroll
      for (int s = 0; s < 2; ++s)
        acc[s * 4 + j] = __builtin_amdgcn_wmma_f32_16x16x32_bf16(
            false, af[s].v, false, bfr.v, (short)0, acc[s * 4 + j], false, false);
    }
    __syncthreads();
  }

  // SiGLU combine: up waves publish through LDS, gate waves store bf16 h
  if (is_up) {
#pragma unroll
    for (int s = 0; s < 2; ++s)
#pragma unroll
      for (int j = 0; j < 4; ++j)
#pragma unroll
        for (int r = 0; r < 8; ++r)
          shX[((m_blk + 4 * s) * 16 + r + 8 * hlf) * 64 + j * 16 + nl] =
              acc[s * 4 + j][r];
  }
  __syncthreads();
  if (!is_up) {
#pragma unroll
    for (int s = 0; s < 2; ++s) {
#pragma unroll
      for (int j = 0; j < 4; ++j) {
#pragma unroll
        for (int r = 0; r < 8; ++r) {
          const int mloc = (m_blk + 4 * s) * 16 + r + 8 * hlf;
          const int floc = j * 16 + nl;
          float g  = acc[s * 4 + j][r];
          float up = shX[mloc * 64 + floc];
          float hv = (g / (1.f + __expf(-g))) * up;  // silu(g)*up
          unsigned ub = __float_as_uint(hv);
          ub = (ub + 0x7FFFu + ((ub >> 16) & 1u)) >> 16;
          hbuf[((size_t)e * TT + mBase + mloc) * FF + fBase + floc] =
              (unsigned short)ub;
        }
      }
    }
  }
}

// ---------------------------------------------------------------------------
// Kernel 3: GEMM2 + weighted combine.  out[t][n] = sum_e rw[t][e]*(h_e@w2_e^T)
// Block 256 thr, tile 128(M) x 64(N); experts looped internally.
// ---------------------------------------------------------------------------
__global__ __launch_bounds__(256) void moe_gemm2_combine(
    const unsigned short* __restrict__ hbuf,
    const unsigned short* __restrict__ w2b,
    const float* __restrict__ rw, float* __restrict__ out) {
  __shared__ __align__(16) unsigned short shA[128 * 40];
  __shared__ __align__(16) unsigned short shB[64 * 40];
  __shared__ float shRW[128 * 8];

  const int tid  = threadIdx.x;
  const int wave = tid >> 5, ln = tid & 31, hlf = ln >> 4, nl = ln & 15;
  const int mBase = blockIdx.x * 128, nBase = blockIdx.y * 64;
  const int m_blk = wave & 3, npair = wave >> 2;

  for (int i = tid; i < 1024; i += 256)
    shRW[i] = rw[(size_t)(mBase + (i >> 3)) * EE + (i & 7)];

  const int lrow = tid >> 2, lq = tid & 3;

  v8f outacc[4] = {};

  for (int e = 0; e < EE; ++e) {
    v8f acc[4] = {};
    const unsigned short* Ah = hbuf + ((size_t)e * TT + mBase) * FF;
    const unsigned short* Bw = w2b  + ((size_t)e * HH + nBase) * FF;

    uint4 ra0 = *(const uint4*)(Ah + (size_t)lrow * FF + lq * 8);
    uint4 ra1 = *(const uint4*)(Ah + (size_t)(lrow + 64) * FF + lq * 8);
    uint4 rb  = *(const uint4*)(Bw + (size_t)lrow * FF + lq * 8);

    for (int kt = 0; kt < FF / 32; ++kt) {
      *(uint4*)&shA[lrow * 40 + lq * 8]        = ra0;
      *(uint4*)&shA[(lrow + 64) * 40 + lq * 8] = ra1;
      *(uint4*)&shB[lrow * 40 + lq * 8]        = rb;
      __syncthreads();

      if (kt + 1 < FF / 32) {
        const int kB = (kt + 1) * 32;
        ra0 = *(const uint4*)(Ah + (size_t)lrow * FF + kB + lq * 8);
        ra1 = *(const uint4*)(Ah + (size_t)(lrow + 64) * FF + kB + lq * 8);
        rb  = *(const uint4*)(Bw + (size_t)lrow * FF + kB + lq * 8);
      }

      FragBF af[2];
#pragma unroll
      for (int s = 0; s < 2; ++s) {
        const unsigned short* arow = &shA[((m_blk + 4 * s) * 16 + nl) * 40];
#pragma unroll
        for (int v = 0; v < 8; ++v) {
          int k0 = hlf * 16 + v * 2;
          af[s].u[v] = *(const unsigned*)(arow + k0);
        }
      }
#pragma unroll
      for (int j = 0; j < 2; ++j) {
        FragBF bfr;
        const unsigned short* brw =
            &shB[(npair * 32 + j * 16 + nl) * 40 + hlf * 16];
#pragma unroll
        for (int v = 0; v < 8; ++v) bfr.u[v] = *(const unsigned*)(brw + 2 * v);
#pragma unroll
        for (int s = 0; s < 2; ++s)
          acc[s * 2 + j] = __builtin_amdgcn_wmma_f32_16x16x32_bf16(
              false, af[s].v, false, bfr.v, (short)0, acc[s * 2 + j], false,
              false);
      }
      __syncthreads();
    }
    // scale this expert's tile by routing weight, accumulate in registers
#pragma unroll
    for (int s = 0; s < 2; ++s)
#pragma unroll
      for (int j = 0; j < 2; ++j)
#pragma unroll
        for (int r = 0; r < 8; ++r) {
          const int mloc = (m_blk + 4 * s) * 16 + r + 8 * hlf;
          outacc[s * 2 + j][r] += shRW[mloc * 8 + e] * acc[s * 2 + j][r];
        }
  }

#pragma unroll
  for (int s = 0; s < 2; ++s)
#pragma unroll
    for (int j = 0; j < 2; ++j)
#pragma unroll
      for (int r = 0; r < 8; ++r) {
        const int mloc = (m_blk + 4 * s) * 16 + r + 8 * hlf;
        const int nloc = npair * 32 + j * 16 + nl;
        out[(size_t)(mBase + mloc) * HH + nBase + nloc] = outacc[s * 2 + j][r];
      }
}

// ---------------------------------------------------------------------------
extern "C" void kernel_launch(void* const* d_in, const int* in_sizes, int n_in,
                              void* d_out, int out_size, void* d_ws, size_t ws_size,
                              hipStream_t stream) {
  const float* X   = (const float*)d_in[0];  // [T,H]
  const float* tw  = (const float*)d_in[1];  // [T,K]
  const int*   ids = (const int*)  d_in[2];  // [T,K]
  const float* w1  = (const float*)d_in[3];  // [E,2F,H]
  const float* w2  = (const float*)d_in[4];  // [E,H,F]
  float* out = (float*)d_out;                // [T,H]

  // workspace layout (16B-aligned sections):
  //   rw  [T,E]      f32    32 KB
  //   Xb  [T,H]      bf16    4 MB
  //   w1b [E,2F,H]   bf16   50.3 MB
  //   w2b [E,H,F]    bf16   25.2 MB
  //   h   [E,T,F]    bf16   12.6 MB
  char* ws = (char*)d_ws;
  float*          rwp  = (float*)ws;
  unsigned short* Xb   = (unsigned short*)(ws + 32768);
  unsigned short* w1b  = (unsigned short*)(ws + 32768 + (size_t)TT * HH * 2);
  unsigned short* w2b  = (unsigned short*)(ws + 32768 + (size_t)TT * HH * 2 +
                                           (size_t)EE * 2 * FF * HH * 2);
  unsigned short* hbuf = (unsigned short*)(ws + 32768 + (size_t)TT * HH * 2 +
                                           (size_t)EE * 2 * FF * HH * 2 +
                                           (size_t)EE * HH * FF * 2);

  const int nX4  = TT * HH / 4;            // 524288
  const int nW14 = EE * 2 * FF * HH / 4;   // 6291456
  const int nW24 = EE * HH * FF / 4;       // 3145728

  moe_route<<<dim3(TT / 256), dim3(256), 0, stream>>>(tw, ids, rwp);
  cvt_f32_bf16_v4<<<dim3(nX4  / 256), dim3(256), 0, stream>>>(X,  Xb,  nX4);
  cvt_f32_bf16_v4<<<dim3(nW14 / 256), dim3(256), 0, stream>>>(w1, w1b, nW14);
  cvt_f32_bf16_v4<<<dim3(nW24 / 256), dim3(256), 0, stream>>>(w2, w2b, nW24);
  moe_gemm1_silu<<<dim3(TT / 128, FF / 64, EE), dim3(256), 0, stream>>>(Xb, w1b, hbuf);
  moe_gemm2_combine<<<dim3(TT / 128, HH / 64), dim3(256), 0, stream>>>(hbuf, w2b, rwp, out);
}
